// RobertaHGAdapterLayer_76836964925851
// MI455X (gfx1250) — matrix operand order repaired and checked by reading, not compiled, not run on hardware
//
#include <hip/hip_runtime.h>
#include <hip/hip_bf16.h>
#include <math.h>

// ---------------- problem constants (match reference) ----------------
constexpr int B_  = 8;
constexpr int S_  = 1024;
constexpr int H_  = 768;
constexpr int NH_ = 12;
constexpr int DH_ = 64;
constexpr int FF_ = 3072;
constexpr int R_  = 64;
constexpr int NE_ = 512;
constexpr int EI_ = 4096;
constexpr int NT_ = 8;
constexpr float EPS_ = 1e-5f;

typedef __attribute__((ext_vector_type(16))) _Float16 v16h;
typedef __attribute__((ext_vector_type(8)))  _Float16 v8h;
typedef __attribute__((ext_vector_type(8)))  float    v8f;
typedef __attribute__((ext_vector_type(4)))  unsigned v4u;
typedef __attribute__((ext_vector_type(4)))  int      v4i;
typedef __attribute__((ext_vector_type(8)))  int      v8i_;

__device__ __forceinline__ float gelu_tanh(float x) {
    const float k0 = 0.7978845608028654f, k1 = 0.044715f;
    return 0.5f * x * (1.0f + tanhf(k0 * (x + k1 * x * x * x)));
}

// ---------------- Tensor Data Mover (gfx1250) helpers ----------------
#if defined(__has_builtin)
#if __has_builtin(__builtin_amdgcn_tensor_load_to_lds)
#define HAVE_TDM_PATH 1
#endif
#endif

#ifdef HAVE_TDM_PATH
// Generic (LDS-aperture) pointer low 32 bits == LDS byte offset (flat LDS map).
__device__ __forceinline__ unsigned lds_off(const void* p) {
    return (unsigned)(unsigned long)(uintptr_t)p;
}
// 2-D tile load: rows x 32 f16 elements, row stride K elements, with LDS
// padding of 4 DWORDs after every 16 DWORDs (64B row) -> 40-half LDS pitch.
// D# layout per CDNA5 ISA.
__device__ __forceinline__ void tdm_load_2d(const _Float16* gsrc, unsigned ldsOff,
                                            int K, int rows) {
    unsigned long ga = (unsigned long)(uintptr_t)gsrc;
    v4u g0;
    g0.x = 1u;                                            // count=1 (valid descriptor)
    g0.y = ldsOff;                                        // lds_addr (bytes)
    g0.z = (unsigned)(ga & 0xFFFFFFFFu);                  // global_addr[31:0]
    g0.w = ((unsigned)(ga >> 32) & 0x01FFFFFFu) | 0x80000000u; // addr[56:32] | type=2
    v8i_ g1;
    g1[0] = (1 << 16) | (1 << 20) | (3 << 22) | (3 << 25); // data_size=2B, pad_en, 16dw interval, 4dw pad
    g1[1] = (K & 0xFFFF) << 16;                           // tensor_dim0[15:0]
    g1[2] = (int)(((unsigned)K >> 16) | ((unsigned)(rows & 0xFFFF) << 16)); // dim0 hi | dim1 lo
    g1[3] = (32 << 16);                                   // dim1 hi=0 | tile_dim0=32
    g1[4] = rows & 0xFFFF;                                // tile_dim1 (tile_dim2=0)
    g1[5] = K;                                            // tensor_dim0_stride[31:0]
    g1[6] = 0;                                            // stride hi | dim1_stride lo
    g1[7] = 0;
    v4i z = {0, 0, 0, 0};
#if __clang_major__ >= 23
    v8i_ z8 = {0, 0, 0, 0, 0, 0, 0, 0};
    __builtin_amdgcn_tensor_load_to_lds(g0, g1, z, z, z8, 0);
#else
    __builtin_amdgcn_tensor_load_to_lds(g0, g1, z, z, 0);
#endif
}
__device__ __forceinline__ void tdm_wait0() {
#if __has_builtin(__builtin_amdgcn_s_wait_tensorcnt)
    __builtin_amdgcn_s_wait_tensorcnt(0);
#else
    asm volatile("s_wait_tensorcnt 0x0" ::: "memory");
#endif
}
#endif // HAVE_TDM_PATH

// ---------------------------------------------------------------------
// WMMA GEMM:  C[b] = act( A[b] (MxK f16) * Bt[b]^T (Bt is NxK f16) * cscale + bias )
// Block 256 threads = 8 waves arranged 4(M) x 2(N) -> 128x64 block tile.
// One wave -> 32x32 tile (2x2 of v_wmma_f32_16x16x32_f16).
// Operands are staged into LDS by the Tensor Data Mover (double-buffered,
// wave 0 issues, syncs on TENSORcnt + workgroup barrier); fragment reads come
// from LDS with a 40-half row pitch (bank-conflict free ds_load_b128).
// Fragment layouts follow CDNA5 ISA 7.12.2 (wave32).
// ---------------------------------------------------------------------
constexpr int LROW = 40;   // 32 payload halfs + 8 pad halfs (4 DWORDs) per K-row

template<bool OUT16, bool GELU, bool BIAS>
__global__ __launch_bounds__(256) void gemm_wmma_k(
    const _Float16* __restrict__ A, const _Float16* __restrict__ Bt,
    const float* __restrict__ bias, void* __restrict__ Cv,
    int M, int N, int K, long sA, long sB, long sC, float cscale)
{
    __shared__ _Float16 lA[2][128 * LROW];   // 20 KB
    __shared__ _Float16 lB[2][64 * LROW];    // 10 KB

    const int tid  = threadIdx.x;
    const int lane = tid & 31;
    const int wid  = tid >> 5;
    const int wm = wid & 3, wn = wid >> 2;
    const long zb = blockIdx.z;
    const _Float16* Ab = A + zb * sA;
    const _Float16* Bb = Bt + zb * sB;
    const int rowA = blockIdx.y * 128;
    const int colB = blockIdx.x * 64;

    const int la = lane & 15;
    const int hi = lane >> 4;          // half-wave select
    const int ka = hi * 8;             // A frag: lanes 16-31 take K+8 of each 16-chunk
    const int kb = hi * 8;             // B frag: lanes 16-31 take K+8 of each 16-chunk

    v8f acc00 = {}, acc01 = {}, acc10 = {}, acc11 = {};

    auto issue_tile = [&](int bi, int kk) {
#ifdef HAVE_TDM_PATH
        if (tid < 32) {   // one wave drives the TDM; EXEC-independent DMA
            tdm_load_2d(Ab + (long)rowA * K + kk, lds_off(&lA[bi][0]), K, 128);
            tdm_load_2d(Bb + (long)colB * K + kk, lds_off(&lB[bi][0]), K, 64);
        }
#else
        // cooperative fallback: 128x32 A tile = 512 x v8h chunks, B = 256 chunks
        for (int q = tid; q < 512; q += 256) {
            int r = q >> 2, c = q & 3;
            *(v8h*)&lA[bi][r * LROW + c * 8] =
                *(const v8h*)(Ab + (long)(rowA + r) * K + kk + c * 8);
        }
        {
            int r = tid >> 2, c = tid & 3;
            *(v8h*)&lB[bi][r * LROW + c * 8] =
                *(const v8h*)(Bb + (long)(colB + r) * K + kk + c * 8);
        }
#endif
    };
    auto wait_tile = [&]() {
#ifdef HAVE_TDM_PATH
        if (tid < 32) tdm_wait0();   // issuing wave drains TENSORcnt
#endif
        __syncthreads();             // release LDS tile to all 8 waves
    };

    issue_tile(0, 0);
    wait_tile();

    int cur = 0;
    for (int kk = 0; kk < K; kk += 32) {
        const int nxt = cur ^ 1;
        if (kk + 32 < K) issue_tile(nxt, kk + 32);   // DMA overlaps WMMA work

        const _Float16* lAc = &lA[cur][0];
        const _Float16* lBc = &lB[cur][0];
        const int mA0 = (wm * 32 +      la) * LROW + ka;
        const int mA1 = (wm * 32 + 16 + la) * LROW + ka;
        const int nB0 = (wn * 32 +      la) * LROW + kb;
        const int nB1 = (wn * 32 + 16 + la) * LROW + kb;

        v8h a0l = *(const v8h*)(lAc + mA0);
        v8h a0h = *(const v8h*)(lAc + mA0 + 16);
        v8h a1l = *(const v8h*)(lAc + mA1);
        v8h a1h = *(const v8h*)(lAc + mA1 + 16);
        v8h b0l = *(const v8h*)(lBc + nB0);
        v8h b0h = *(const v8h*)(lBc + nB0 + 16);
        v8h b1l = *(const v8h*)(lBc + nB1);
        v8h b1h = *(const v8h*)(lBc + nB1 + 16);
        v16h a0 = __builtin_shufflevector(a0l, a0h, 0,1,2,3,4,5,6,7,8,9,10,11,12,13,14,15);
        v16h a1 = __builtin_shufflevector(a1l, a1h, 0,1,2,3,4,5,6,7,8,9,10,11,12,13,14,15);
        v16h b0 = __builtin_shufflevector(b0l, b0h, 0,1,2,3,4,5,6,7,8,9,10,11,12,13,14,15);
        v16h b1 = __builtin_shufflevector(b1l, b1h, 0,1,2,3,4,5,6,7,8,9,10,11,12,13,14,15);

        acc00 = __builtin_amdgcn_wmma_f32_16x16x32_f16(false, a0, false, b0, (short)0, acc00, false, false);
        acc01 = __builtin_amdgcn_wmma_f32_16x16x32_f16(false, a0, false, b1, (short)0, acc01, false, false);
        acc10 = __builtin_amdgcn_wmma_f32_16x16x32_f16(false, a1, false, b0, (short)0, acc10, false, false);
        acc11 = __builtin_amdgcn_wmma_f32_16x16x32_f16(false, a1, false, b1, (short)0, acc11, false, false);

        __syncthreads();                      // all waves done reading cur
        if (kk + 32 < K) wait_tile();         // nxt is resident -> becomes cur
        cur = nxt;
    }

    const int m0 = rowA + wm * 32;
    const int n0 = colB + wn * 32;
    // C/D layout: VGPR r -> row m0+16*mi + hi*8 + r ; col = n0+16*nj + (lane&15)
    auto store_tile = [&](const v8f& acc, int mi, int nj) {
        const int col = n0 + 16 * nj + la;
        const float bs = BIAS ? bias[col] : 0.0f;
#pragma unroll
        for (int r = 0; r < 8; ++r) {
            const int row = m0 + 16 * mi + hi * 8 + r;
            float v = acc[r] * cscale + bs;
            if (GELU) v = gelu_tanh(v);
            const long idx = (long)row * N + col;
            if (OUT16) ((_Float16*)Cv)[zb * sC + idx] = (_Float16)v;
            else       ((float*)Cv)[zb * sC + idx] = v;
        }
    };
    store_tile(acc00, 0, 0); store_tile(acc01, 0, 1);
    store_tile(acc10, 1, 0); store_tile(acc11, 1, 1);
}

// ---------------- elementwise / layout kernels ----------------
__global__ void cvt_f32_f16_k(const float* __restrict__ in, _Float16* __restrict__ out, long n) {
    long i = (long)blockIdx.x * blockDim.x + threadIdx.x;
    if (i < n) out[i] = (_Float16)in[i];
}

// W [K,N] f32 -> Wt [N,K] f16 (so WMMA B fragments are contiguous)
__global__ void cvtT_f32_f16_k(const float* __restrict__ W, _Float16* __restrict__ Wt, int K, int N) {
    long i = (long)blockIdx.x * blockDim.x + threadIdx.x;
    if (i < (long)K * N) {
        int k = (int)(i / N), n = (int)(i % N);
        Wt[(long)n * K + k] = (_Float16)W[i];
    }
}

// [B,S,H] f16 -> [B,NH,S,DH] f16
__global__ void pack_qk_k(const _Float16* __restrict__ in, _Float16* __restrict__ out) {
    long t = (long)blockIdx.x * blockDim.x + threadIdx.x;
    if (t >= (long)B_ * S_ * H_) return;
    int b = (int)(t / ((long)S_ * H_));
    long rem = t % ((long)S_ * H_);
    int s = (int)(rem / H_), c = (int)(rem % H_);
    int h = c / DH_, d = c % DH_;
    out[(((long)b * NH_ + h) * S_ + s) * DH_ + d] = in[t];
}

// [B,S,H] f16 -> [B,NH,DH,S] f16 (V transposed per head, = Bt for probs@V)
__global__ void pack_vT_k(const _Float16* __restrict__ in, _Float16* __restrict__ out) {
    long t = (long)blockIdx.x * blockDim.x + threadIdx.x;
    if (t >= (long)B_ * S_ * H_) return;
    int b = (int)(t / ((long)S_ * H_));
    long rem = t % ((long)S_ * H_);
    int s = (int)(rem / H_), c = (int)(rem % H_);
    int h = c / DH_, d = c % DH_;
    out[(((long)b * NH_ + h) * DH_ + d) * S_ + s] = in[t];
}

// [B,NH,S,DH] f16 -> [B,S,H] f16
__global__ void unpack_ctx_k(const _Float16* __restrict__ in, _Float16* __restrict__ out) {
    long t = (long)blockIdx.x * blockDim.x + threadIdx.x;
    if (t >= (long)B_ * S_ * H_) return;
    int b = (int)(t / ((long)S_ * H_));
    long rem = t % ((long)S_ * H_);
    int s = (int)(rem / H_), c = (int)(rem % H_);
    int h = c / DH_, d = c % DH_;
    out[t] = in[(((long)b * NH_ + h) * S_ + s) * DH_ + d];
}

// row-wise softmax in place, f16 storage / fp32 math, row length S_=1024
__global__ __launch_bounds__(256) void softmax_k(_Float16* __restrict__ p) {
    __shared__ float red[256];
    const long row = blockIdx.x;
    _Float16* pr = p + row * (long)S_;
    const int t = threadIdx.x;
    float vals[4];
    float m = -1e30f;
#pragma unroll
    for (int j = 0; j < 4; ++j) { vals[j] = (float)pr[t + 256 * j]; m = fmaxf(m, vals[j]); }
    red[t] = m; __syncthreads();
    for (int k = 128; k > 0; k >>= 1) { if (t < k) red[t] = fmaxf(red[t], red[t + k]); __syncthreads(); }
    m = red[0]; __syncthreads();
    float sum = 0.f;
#pragma unroll
    for (int j = 0; j < 4; ++j) { vals[j] = __expf(vals[j] - m); sum += vals[j]; }
    red[t] = sum; __syncthreads();
    for (int k = 128; k > 0; k >>= 1) { if (t < k) red[t] += red[t + k]; __syncthreads(); }
    const float inv = 1.0f / red[0];
#pragma unroll
    for (int j = 0; j < 4; ++j) pr[t + 256 * j] = (_Float16)(vals[j] * inv);
}

// out = LN(a + b + c) ; optional f16 mirror
__global__ __launch_bounds__(256) void residual_ln_k(
    const float* __restrict__ a, const float* __restrict__ b2, const float* __restrict__ c,
    const float* __restrict__ g, const float* __restrict__ be,
    float* __restrict__ out, _Float16* __restrict__ outh)
{
    __shared__ float s1[256], s2[256];
    const long row = blockIdx.x;
    const float *ar = a + row * H_, *br = b2 + row * H_, *cr = c + row * H_;
    const int t = threadIdx.x;
    float vals[3]; float s = 0.f, ss = 0.f;
#pragma unroll
    for (int j = 0; j < 3; ++j) {
        int idx = t + 256 * j;
        float v = ar[idx] + br[idx] + cr[idx];
        vals[j] = v; s += v; ss += v * v;
    }
    s1[t] = s; s2[t] = ss; __syncthreads();
    for (int k = 128; k > 0; k >>= 1) {
        if (t < k) { s1[t] += s1[t + k]; s2[t] += s2[t + k]; }
        __syncthreads();
    }
    const float mean = s1[0] / H_;
    const float var  = s2[0] / H_ - mean * mean;
    const float inv  = rsqrtf(var + EPS_);
#pragma unroll
    for (int j = 0; j < 3; ++j) {
        int idx = t + 256 * j;
        float y = (vals[j] - mean) * inv * g[idx] + be[idx];
        out[row * H_ + idx] = y;
        if (outh) outh[row * H_ + idx] = (_Float16)y;
    }
}

// ---------------- hypergraph adapter kernels ----------------
__global__ void fill0_k(float* __restrict__ p, long n) {
    long i = (long)blockIdx.x * blockDim.x + threadIdx.x;
    if (i < n) p[i] = 0.0f;
}

__global__ void gather_edge_k(const float* __restrict__ d, const int* __restrict__ nidx,
                              const int* __restrict__ eidx, float* __restrict__ esum,
                              float* __restrict__ ecnt) {
    long t = (long)blockIdx.x * blockDim.x + threadIdx.x;
    if (t >= (long)B_ * EI_) return;
    int b = (int)(t / EI_);
    int n = nidx[t], e = eidx[t];
    const float* src = d + ((long)b * S_ + n) * R_;
    float* dst = esum + ((long)b * NE_ + e) * R_;
    for (int r = 0; r < R_; ++r) atomicAdd(&dst[r], src[r]);
    atomicAdd(&ecnt[(long)b * NE_ + e], 1.0f);
}

__global__ void edge_final_k(const float* __restrict__ esum, const float* __restrict__ ecnt,
                             const int* __restrict__ types, const float* __restrict__ temb,
                             float* __restrict__ e) {
    long t = (long)blockIdx.x * blockDim.x + threadIdx.x;
    if (t >= (long)B_ * NE_ * R_) return;
    long be = t / R_; int r = (int)(t % R_);
    float cnt = fmaxf(ecnt[be], 1.0f);
    int ty = types[be];
    e[t] = esum[t] / cnt + temb[(long)ty * R_ + r];
}

__global__ void scatter_node_k(const float* __restrict__ e, const int* __restrict__ nidx,
                               const int* __restrict__ eidx, float* __restrict__ nsum,
                               float* __restrict__ ncnt) {
    long t = (long)blockIdx.x * blockDim.x + threadIdx.x;
    if (t >= (long)B_ * EI_) return;
    int b = (int)(t / EI_);
    int n = nidx[t], ed = eidx[t];
    const float* src = e + ((long)b * NE_ + ed) * R_;
    float* dst = nsum + ((long)b * S_ + n) * R_;
    for (int r = 0; r < R_; ++r) atomicAdd(&dst[r], src[r]);
    atomicAdd(&ncnt[(long)b * S_ + n], 1.0f);
}

__global__ void node_final_k(const float* __restrict__ nsum, const float* __restrict__ ncnt,
                             _Float16* __restrict__ uh) {
    long t = (long)blockIdx.x * blockDim.x + threadIdx.x;
    if (t >= (long)B_ * S_ * R_) return;
    long bs = t / R_;
    float cnt = fmaxf(ncnt[bs], 1.0f);
    uh[t] = (_Float16)(nsum[t] / cnt);
}

// ---------------------------------------------------------------------
extern "C" void kernel_launch(void* const* d_in, const int* in_sizes, int n_in,
                              void* d_out, int out_size, void* d_ws, size_t ws_size,
                              hipStream_t stream)
{
    (void)in_sizes; (void)n_in; (void)out_size; (void)ws_size;
    const float* x    = (const float*)d_in[0];
    const int*   nidx = (const int*)d_in[1];
    const int*   eidx = (const int*)d_in[2];
    const int*   etyp = (const int*)d_in[3];
    const float* Wq = (const float*)d_in[4];  const float* bq = (const float*)d_in[5];
    const float* Wk = (const float*)d_in[6];  const float* bk = (const float*)d_in[7];
    const float* Wv = (const float*)d_in[8];  const float* bv = (const float*)d_in[9];
    const float* Wo = (const float*)d_in[10]; const float* bo = (const float*)d_in[11];
    const float* ln1g = (const float*)d_in[12]; const float* ln1b = (const float*)d_in[13];
    const float* a1dw = (const float*)d_in[14]; const float* a1db = (const float*)d_in[15];
    const float* a1uw = (const float*)d_in[16]; const float* a1ub = (const float*)d_in[17];
    const float* a1te = (const float*)d_in[18];
    const float* Wi  = (const float*)d_in[19]; const float* bi  = (const float*)d_in[20];
    const float* Wd2 = (const float*)d_in[21]; const float* bd2 = (const float*)d_in[22];
    const float* ln2g = (const float*)d_in[23]; const float* ln2b = (const float*)d_in[24];
    const float* a2dw = (const float*)d_in[25]; const float* a2db = (const float*)d_in[26];
    const float* a2uw = (const float*)d_in[27]; const float* a2ub = (const float*)d_in[28];
    const float* a2te = (const float*)d_in[29];

    // ---- workspace bump allocator (256B aligned) ----
    size_t off = 0;
    auto alloc = [&](size_t bytes) -> void* {
        void* p = (char*)d_ws + off;
        off += (bytes + 255) & ~(size_t)255;
        return p;
    };
    const long nBSH = (long)B_ * S_ * H_;

    _Float16* WqT   = (_Float16*)alloc((size_t)H_ * H_ * 2);
    _Float16* WkT   = (_Float16*)alloc((size_t)H_ * H_ * 2);
    _Float16* WvT   = (_Float16*)alloc((size_t)H_ * H_ * 2);
    _Float16* WoT   = (_Float16*)alloc((size_t)H_ * H_ * 2);
    _Float16* WiT   = (_Float16*)alloc((size_t)H_ * FF_ * 2);
    _Float16* Wd2T  = (_Float16*)alloc((size_t)H_ * FF_ * 2);
    _Float16* a1dwT = (_Float16*)alloc((size_t)H_ * R_ * 2);
    _Float16* a1uwT = (_Float16*)alloc((size_t)H_ * R_ * 2);
    _Float16* a2dwT = (_Float16*)alloc((size_t)H_ * R_ * 2);
    _Float16* a2uwT = (_Float16*)alloc((size_t)H_ * R_ * 2);
    _Float16* xh    = (_Float16*)alloc((size_t)nBSH * 2);
    _Float16* qtmp  = (_Float16*)alloc((size_t)nBSH * 2);
    _Float16* ktmp  = (_Float16*)alloc((size_t)nBSH * 2);
    _Float16* vtmp  = (_Float16*)alloc((size_t)nBSH * 2);
    _Float16* qh    = (_Float16*)alloc((size_t)nBSH * 2);
    _Float16* kh    = (_Float16*)alloc((size_t)nBSH * 2);
    _Float16* vt    = (_Float16*)alloc((size_t)nBSH * 2);
    _Float16* probs = (_Float16*)alloc((size_t)B_ * NH_ * S_ * S_ * 2);
    _Float16* ctx   = (_Float16*)alloc((size_t)nBSH * 2);
    _Float16* xctx  = (_Float16*)alloc((size_t)nBSH * 2);
    float*    attn_dense = (float*)alloc((size_t)nBSH * 4);
    _Float16* adh   = (_Float16*)alloc((size_t)nBSH * 2);
    float*    dbuf  = (float*)alloc((size_t)B_ * S_ * R_ * 4);
    float*    esum  = (float*)alloc((size_t)B_ * NE_ * R_ * 4);
    float*    ecnt  = (float*)alloc((size_t)B_ * NE_ * 4);
    float*    ebuf  = (float*)alloc((size_t)B_ * NE_ * R_ * 4);
    float*    nsum  = (float*)alloc((size_t)B_ * S_ * R_ * 4);
    float*    ncnt  = (float*)alloc((size_t)B_ * S_ * 4);
    _Float16* uh    = (_Float16*)alloc((size_t)B_ * S_ * R_ * 2);
    float*    adbuf = (float*)alloc((size_t)nBSH * 4);
    float*    attn_out = (float*)alloc((size_t)nBSH * 4);
    _Float16* aoh   = (_Float16*)alloc((size_t)nBSH * 2);
    _Float16* inter = (_Float16*)alloc((size_t)B_ * S_ * FF_ * 2);
    float*    ffbuf = (float*)alloc((size_t)nBSH * 4);
    _Float16* ffh   = (_Float16*)alloc((size_t)nBSH * 2);

    auto EW = [&](long n) { return dim3((unsigned)((n + 255) / 256), 1, 1); };

    auto gemm = [&](const _Float16* A, const _Float16* Bt, const float* bias, void* C,
                    int M, int N, int K, int batch, long sA, long sB, long sC,
                    float scale, bool out16, bool gelu) {
        dim3 g(N / 64, M / 128, batch), b(256, 1, 1);
        const bool hb = (bias != nullptr);
        if (out16) {
            if (gelu) {
                if (hb) gemm_wmma_k<true, true, true ><<<g, b, 0, stream>>>(A, Bt, bias, C, M, N, K, sA, sB, sC, scale);
                else    gemm_wmma_k<true, true, false><<<g, b, 0, stream>>>(A, Bt, bias, C, M, N, K, sA, sB, sC, scale);
            } else {
                if (hb) gemm_wmma_k<true, false, true ><<<g, b, 0, stream>>>(A, Bt, bias, C, M, N, K, sA, sB, sC, scale);
                else    gemm_wmma_k<true, false, false><<<g, b, 0, stream>>>(A, Bt, bias, C, M, N, K, sA, sB, sC, scale);
            }
        } else {
            if (gelu) {
                if (hb) gemm_wmma_k<false, true, true ><<<g, b, 0, stream>>>(A, Bt, bias, C, M, N, K, sA, sB, sC, scale);
                else    gemm_wmma_k<false, true, false><<<g, b, 0, stream>>>(A, Bt, bias, C, M, N, K, sA, sB, sC, scale);
            } else {
                if (hb) gemm_wmma_k<false, false, true ><<<g, b, 0, stream>>>(A, Bt, bias, C, M, N, K, sA, sB, sC, scale);
                else    gemm_wmma_k<false, false, false><<<g, b, 0, stream>>>(A, Bt, bias, C, M, N, K, sA, sB, sC, scale);
            }
        }
    };

    // ---- weight conversions (fp32 [K,N] -> f16 [N,K]) ----
    cvtT_f32_f16_k<<<EW((long)H_ * H_), 256, 0, stream>>>(Wq,  WqT,  H_, H_);
    cvtT_f32_f16_k<<<EW((long)H_ * H_), 256, 0, stream>>>(Wk,  WkT,  H_, H_);
    cvtT_f32_f16_k<<<EW((long)H_ * H_), 256, 0, stream>>>(Wv,  WvT,  H_, H_);
    cvtT_f32_f16_k<<<EW((long)H_ * H_), 256, 0, stream>>>(Wo,  WoT,  H_, H_);
    cvtT_f32_f16_k<<<EW((long)H_ * FF_), 256, 0, stream>>>(Wi,  WiT,  H_, FF_);
    cvtT_f32_f16_k<<<EW((long)H_ * FF_), 256, 0, stream>>>(Wd2, Wd2T, FF_, H_);
    cvtT_f32_f16_k<<<EW((long)H_ * R_), 256, 0, stream>>>(a1dw, a1dwT, H_, R_);
    cvtT_f32_f16_k<<<EW((long)H_ * R_), 256, 0, stream>>>(a1uw, a1uwT, R_, H_);
    cvtT_f32_f16_k<<<EW((long)H_ * R_), 256, 0, stream>>>(a2dw, a2dwT, H_, R_);
    cvtT_f32_f16_k<<<EW((long)H_ * R_), 256, 0, stream>>>(a2uw, a2uwT, R_, H_);
    cvt_f32_f16_k<<<EW(nBSH), 256, 0, stream>>>(x, xh, nBSH);

    // ---- QKV projections ----
    gemm(xh, WqT, bq, qtmp, B_ * S_, H_, H_, 1, 0, 0, 0, 1.f, true, false);
    gemm(xh, WkT, bk, ktmp, B_ * S_, H_, H_, 1, 0, 0, 0, 1.f, true, false);
    gemm(xh, WvT, bv, vtmp, B_ * S_, H_, H_, 1, 0, 0, 0, 1.f, true, false);
    pack_qk_k<<<EW(nBSH), 256, 0, stream>>>(qtmp, qh);
    pack_qk_k<<<EW(nBSH), 256, 0, stream>>>(ktmp, kh);
    pack_vT_k<<<EW(nBSH), 256, 0, stream>>>(vtmp, vt);

    // ---- attention: scores (96 batched GEMMs), softmax, context ----
    gemm(qh, kh, nullptr, probs, S_, S_, DH_, B_ * NH_,
         (long)S_ * DH_, (long)S_ * DH_, (long)S_ * S_, 0.125f, true, false);
    softmax_k<<<dim3(B_ * NH_ * S_), 256, 0, stream>>>(probs);
    gemm(probs, vt, nullptr, ctx, S_, DH_, S_, B_ * NH_,
         (long)S_ * S_, (long)DH_ * S_, (long)S_ * DH_, 1.f, true, false);
    unpack_ctx_k<<<EW(nBSH), 256, 0, stream>>>(ctx, xctx);
    gemm(xctx, WoT, bo, attn_dense, B_ * S_, H_, H_, 1, 0, 0, 0, 1.f, false, false);
    cvt_f32_f16_k<<<EW(nBSH), 256, 0, stream>>>(attn_dense, adh, nBSH);

    // ---- hypergraph adapter (shared between both sites) ----
    auto adapter = [&](const _Float16* inh, const _Float16* dwT, const float* db,
                       const _Float16* uwT, const float* ub, const float* temb,
                       float* out) {
        gemm(inh, dwT, db, dbuf, B_ * S_, R_, H_, 1, 0, 0, 0, 1.f, false, true);
        fill0_k<<<EW((long)B_ * NE_ * R_), 256, 0, stream>>>(esum, (long)B_ * NE_ * R_);
        fill0_k<<<EW((long)B_ * NE_),       256, 0, stream>>>(ecnt, (long)B_ * NE_);
        fill0_k<<<EW((long)B_ * S_ * R_),  256, 0, stream>>>(nsum, (long)B_ * S_ * R_);
        fill0_k<<<EW((long)B_ * S_),        256, 0, stream>>>(ncnt, (long)B_ * S_);
        gather_edge_k<<<EW((long)B_ * EI_), 256, 0, stream>>>(dbuf, nidx, eidx, esum, ecnt);
        edge_final_k<<<EW((long)B_ * NE_ * R_), 256, 0, stream>>>(esum, ecnt, etyp, temb, ebuf);
        scatter_node_k<<<EW((long)B_ * EI_), 256, 0, stream>>>(ebuf, nidx, eidx, nsum, ncnt);
        node_final_k<<<EW((long)B_ * S_ * R_), 256, 0, stream>>>(nsum, ncnt, uh);
        gemm(uh, uwT, ub, out, B_ * S_, H_, R_, 1, 0, 0, 0, 1.f, false, false);
    };

    adapter(adh, a1dwT, a1db, a1uwT, a1ub, a1te, adbuf);
    residual_ln_k<<<dim3(B_ * S_), 256, 0, stream>>>(attn_dense, adbuf, x, ln1g, ln1b, attn_out, aoh);

    // ---- FFN ----
    gemm(aoh, WiT, bi, inter, B_ * S_, FF_, H_, 1, 0, 0, 0, 1.f, true, true);
    gemm(inter, Wd2T, bd2, ffbuf, B_ * S_, H_, FF_, 1, 0, 0, 0, 1.f, false, false);
    cvt_f32_f16_k<<<EW(nBSH), 256, 0, stream>>>(ffbuf, ffh, nBSH);

    // ---- adapter 2 + final LN -> d_out ----
    adapter(ffh, a2dwT, a2db, a2uwT, a2ub, a2te, adbuf);
    residual_ln_k<<<dim3(B_ * S_), 256, 0, stream>>>(ffbuf, adbuf, attn_out, ln2g, ln2b,
                                                     (float*)d_out, (_Float16*)nullptr);
}
